// MultiheadAttention_43327630082647
// MI455X (gfx1250) — matrix operand order repaired and checked by reading, not compiled, not run on hardware
//
#include <hip/hip_runtime.h>
#include <hip/hip_bf16.h>

typedef __attribute__((ext_vector_type(16))) _Float16 v16h;
typedef __attribute__((ext_vector_type(8)))  _Float16 v8h;
typedef __attribute__((ext_vector_type(8)))  float    v8f;
typedef __attribute__((ext_vector_type(4)))  float    v4f;

#define B_  4
#define S_  2048
#define D_  1024
#define H_  16
#define DH_ 64
#define M_  (B_*S_)                 // 8192 rows per projection GEMM
#define MN_ ((size_t)M_*(size_t)D_) // elements per projection output (8M)
#define DD_ ((size_t)D_*(size_t)D_) // 1M elements per weight matrix

static __device__ __forceinline__ v8f wmma_f16(v16h a, v16h b, v8f c) {
  // D = A(16x32 f16) x B(32x16 f16) + C(16x16 f32)
  return __builtin_amdgcn_wmma_f32_16x16x32_f16(false, a, false, b,
                                                (short)0, c, false, false);
}

// ---------------------------------------------------------------------------
// K1: fp32 -> fp16 weight conversion
// ---------------------------------------------------------------------------
__global__ __launch_bounds__(256) void cvt_f32_f16(const float* __restrict__ src,
                                                   _Float16* __restrict__ dst, int n) {
  int i = blockIdx.x * blockDim.x + threadIdx.x;
  if (i < n) dst[i] = (_Float16)src[i];
}

// ---------------------------------------------------------------------------
// K2: Y[8192,1024](f16) = X[8192,1024](f32) @ W^T + bias   (W f16 [1024,1024])
// Block = 8 waves, each owning a 32-row slice of a 256x128 block tile.
// Wave tile 32x128: acc[2][8]; A reused across 8 n-tiles (24 B + 2 cvt-halves
// per WMMA); all waves share the same 128 weight columns -> L0 reuse.
// ---------------------------------------------------------------------------
__global__ __launch_bounds__(256) void qkv_gemm(const float* __restrict__ X,
                                                const _Float16* __restrict__ W,
                                                const float* __restrict__ bias,
                                                _Float16* __restrict__ Y) {
  const int lane = threadIdx.x & 31;
  const int w    = threadIdx.x >> 5;
  const int lh   = lane & 15;
  const int lg   = lane >> 4;
  const int mbase = blockIdx.x * 256 + w * 32;
  const int nbase = blockIdx.y * 128;

  v8f acc[2][8];
#pragma unroll
  for (int mi = 0; mi < 2; ++mi)
#pragma unroll
    for (int ni = 0; ni < 8; ++ni) { v8f z = {}; acc[mi][ni] = z; }

  for (int k = 0; k < D_; k += 32) {
    // A fragments: lane l holds row (l&15); K-range (l>>4)*16
    v16h a[2];
#pragma unroll
    for (int mi = 0; mi < 2; ++mi) {
      const float* pa = X + (size_t)(mbase + mi * 16 + lh) * D_ + k + lg * 16;
      v4f f0 = *(const v4f*)(pa);
      v4f f1 = *(const v4f*)(pa + 4);
      v4f f2 = *(const v4f*)(pa + 8);
      v4f f3 = *(const v4f*)(pa + 12);
#pragma unroll
      for (int i = 0; i < 4; ++i) {
        a[mi][i]      = (_Float16)f0[i];
        a[mi][4 + i]  = (_Float16)f1[i];
        a[mi][8 + i]  = (_Float16)f2[i];
        a[mi][12 + i] = (_Float16)f3[i];
      }
    }
    // B fragments: lane l holds column (l&15) of B = row of W, K-range (l>>4)*16
    v16h b[8];
#pragma unroll
    for (int ni = 0; ni < 8; ++ni)
      b[ni] = *(const v16h*)(W + (size_t)(nbase + ni * 16 + lh) * D_ + k + lg * 16);

#pragma unroll
    for (int mi = 0; mi < 2; ++mi)
#pragma unroll
      for (int ni = 0; ni < 8; ++ni)
        acc[mi][ni] = wmma_f16(a[mi], b[ni], acc[mi][ni]);
  }

  // Epilogue: C layout -> element r is (row mbase+mi*16+r+8*lg, col nbase+ni*16+lh)
#pragma unroll
  for (int ni = 0; ni < 8; ++ni) {
    const int col = nbase + ni * 16 + lh;
    const float bv = bias[col];
#pragma unroll
    for (int mi = 0; mi < 2; ++mi)
#pragma unroll
      for (int r = 0; r < 8; ++r) {
        const int row = mbase + mi * 16 + r + 8 * lg;
        Y[(size_t)row * D_ + col] = (_Float16)(acc[mi][ni][r] + bv);
      }
  }
}

// ---------------------------------------------------------------------------
// K3: per-position head attention. One wave per (b,s) position.
//   scores = Q@K^T (2 WMMAs), softmax over t, +mask (AFTER softmax),
//   out = attn@V (4 WMMAs, K=16 zero-padded to 32), *0.125, store f16.
// ---------------------------------------------------------------------------
__global__ __launch_bounds__(256) void attn_kernel(const _Float16* __restrict__ QKV,
                                                   const float* __restrict__ mask,
                                                   _Float16* __restrict__ Oh) {
  __shared__ _Float16 ldsA[8][16 * 16];   // attn, row-major [h][t]
  __shared__ _Float16 ldsVT[8][64 * 16];  // V^T,  [d][t]

  const int lane = threadIdx.x & 31;
  const int w    = threadIdx.x >> 5;
  const int lh   = lane & 15;
  const int lg   = lane >> 4;
  const int p    = blockIdx.x * 8 + w;  // position in [0, 8192)

  const _Float16* Qp = QKV + (size_t)p * D_;
  const _Float16* Kp = QKV + MN_ + (size_t)p * D_;
  const _Float16* Vp = QKV + 2 * MN_ + (size_t)p * D_;

  // scores[h,t] = sum_d Q[h,d] K[t,d]  (A=Q 16x64, B=K^T 64x16)
  v8f sc = {};
  for (int kk = 0; kk < DH_; kk += 32) {
    v16h a;
    {
      const _Float16* pa = Qp + lh * DH_ + kk + lg * 16;
      v8h lo = *(const v8h*)pa;
      v8h hi = *(const v8h*)(pa + 8);
#pragma unroll
      for (int i = 0; i < 8; ++i) { a[i] = lo[i]; a[i + 8] = hi[i]; }
    }
    v16h b = *(const v16h*)(Kp + lh * DH_ + kk + lg * 16);  // column t = lh
    sc = wmma_f16(a, b, sc);
  }

  // softmax over t: each row h lives across one 16-lane half in one VGPR
  float attnv[8];
#pragma unroll
  for (int r = 0; r < 8; ++r) {
    float x = sc[r];
    float mx = x;
    for (int d = 1; d < 16; d <<= 1) mx = fmaxf(mx, __shfl_xor(mx, d, 32));
    float e = __expf(x - mx);
    float s = e;
    for (int d = 1; d < 16; d <<= 1) s += __shfl_xor(s, d, 32);
    const int h = r + 8 * lg;
    attnv[r] = e / s + mask[h * 16 + lh];  // faithful quirk: mask after softmax
  }

  // stage attn and V^T through LDS to rebuild WMMA fragments
#pragma unroll
  for (int r = 0; r < 8; ++r)
    ldsA[w][(r + 8 * lg) * 16 + lh] = (_Float16)attnv[r];
  {
    const _Float16* pv = Vp + lh * DH_ + lg * 32;  // lane: row t=lh, d-range lg*32..+31
#pragma unroll
    for (int c = 0; c < 4; ++c) {
      v8h vv = *(const v8h*)(pv + c * 8);
#pragma unroll
      for (int i = 0; i < 8; ++i)
        ldsVT[w][(lg * 32 + c * 8 + i) * 16 + lh] = vv[i];
    }
  }
  __syncthreads();

  // A = attn (16x16, K padded to 32 with zeros)
  v16h a;
  {
    v8h lo = *(const v8h*)(&ldsA[w][lh * 16 + lg * 8]);
#pragma unroll
    for (int i = 0; i < 8; ++i) { a[i] = lo[i]; a[i + 8] = (_Float16)0.f; }
  }
#pragma unroll
  for (int ni = 0; ni < 4; ++ni) {
    v16h b;
#pragma unroll
    for (int i = 0; i < 8; ++i) { b[i] = ldsVT[w][(ni * 16 + lh) * 16 + lg * 8 + i]; b[i + 8] = (_Float16)0.f; }
    v8f o = {};
    o = wmma_f16(a, b, o);
#pragma unroll
    for (int r = 0; r < 8; ++r) {
      const int h = r + 8 * lg;
      const int d = ni * 16 + lh;
      // faithful quirk: 1/sqrt(64) applied AFTER PV; fold into f16 store
      Oh[(size_t)p * D_ + h * DH_ + d] = (_Float16)(o[r] * 0.125f);
    }
  }
}

// ---------------------------------------------------------------------------
// K4: out[131072,64](f32) = Oh[131072,64](f16) @ Wp^T + bias
// ---------------------------------------------------------------------------
__global__ __launch_bounds__(256) void proj_gemm(const _Float16* __restrict__ Oh,
                                                 const _Float16* __restrict__ Wp,
                                                 const float* __restrict__ bias,
                                                 float* __restrict__ out) {
  const int lane = threadIdx.x & 31;
  const int w    = threadIdx.x >> 5;
  const int lh   = lane & 15;
  const int lg   = lane >> 4;
  const int rbase = blockIdx.x * 256 + w * 32;

  // preload all B fragments (Wp is 8KB, L2/L0 resident)
  v16h bfr[2][4];
#pragma unroll
  for (int kt = 0; kt < 2; ++kt)
#pragma unroll
    for (int ni = 0; ni < 4; ++ni)
      bfr[kt][ni] = *(const v16h*)(Wp + (size_t)(ni * 16 + lh) * DH_ + kt * 32 + lg * 16);

  v8f acc[2][4];
#pragma unroll
  for (int mi = 0; mi < 2; ++mi)
#pragma unroll
    for (int ni = 0; ni < 4; ++ni) { v8f z = {}; acc[mi][ni] = z; }

#pragma unroll
  for (int kt = 0; kt < 2; ++kt) {
#pragma unroll
    for (int mi = 0; mi < 2; ++mi) {
      v16h a;
      {
        const _Float16* pa = Oh + (size_t)(rbase + mi * 16 + lh) * DH_ + kt * 32 + lg * 16;
        v8h lo = *(const v8h*)pa;
        v8h hi = *(const v8h*)(pa + 8);
#pragma unroll
        for (int i = 0; i < 8; ++i) { a[i] = lo[i]; a[i + 8] = hi[i]; }
      }
#pragma unroll
      for (int ni = 0; ni < 4; ++ni)
        acc[mi][ni] = wmma_f16(a, bfr[kt][ni], acc[mi][ni]);
    }
  }

#pragma unroll
  for (int ni = 0; ni < 4; ++ni) {
    const float bv = bias[ni * 16 + lh];
#pragma unroll
    for (int mi = 0; mi < 2; ++mi)
#pragma unroll
      for (int r = 0; r < 8; ++r) {
        const int row = rbase + mi * 16 + r + 8 * lg;
        out[(size_t)row * DH_ + ni * 16 + lh] = acc[mi][ni][r] + bv;
      }
  }
}

// ---------------------------------------------------------------------------
extern "C" void kernel_launch(void* const* d_in, const int* in_sizes, int n_in,
                              void* d_out, int out_size, void* d_ws, size_t ws_size,
                              hipStream_t stream) {
  (void)in_sizes; (void)n_in; (void)out_size; (void)ws_size;

  const float* target = (const float*)d_in[0];
  const float* source = (const float*)d_in[1];
  const float* memory = (const float*)d_in[2];
  const float* maskp  = (const float*)d_in[3];
  const float* WQ_w = (const float*)d_in[4];
  const float* WQ_b = (const float*)d_in[5];
  const float* WK_w = (const float*)d_in[6];
  const float* WK_b = (const float*)d_in[7];
  const float* WV_w = (const float*)d_in[8];
  const float* WV_b = (const float*)d_in[9];
  const float* Wp_w = (const float*)d_in[10];
  const float* Wp_b = (const float*)d_in[11];

  // workspace carve-out (~73.4 MB, all offsets 32B-aligned)
  char* ws = (char*)d_ws;
  _Float16* Wh    = (_Float16*)ws; ws += 3 * DD_ * sizeof(_Float16);
  _Float16* Wph   = (_Float16*)ws; ws += (size_t)DH_ * DH_ * sizeof(_Float16);
  _Float16* QKVh  = (_Float16*)ws; ws += 3 * MN_ * sizeof(_Float16);
  _Float16* Ohalf = (_Float16*)ws; ws += MN_ * sizeof(_Float16);

  // K1: weight conversion
  cvt_f32_f16<<<(int)((DD_ + 255) / 256), 256, 0, stream>>>(WQ_w, Wh, (int)DD_);
  cvt_f32_f16<<<(int)((DD_ + 255) / 256), 256, 0, stream>>>(WK_w, Wh + DD_, (int)DD_);
  cvt_f32_f16<<<(int)((DD_ + 255) / 256), 256, 0, stream>>>(WV_w, Wh + 2 * DD_, (int)DD_);
  cvt_f32_f16<<<(DH_ * DH_ + 255) / 256, 256, 0, stream>>>(Wp_w, Wph, DH_ * DH_);

  // K2: Q/K/V projections (M=8192, N=K=1024 each), 256x128 block tiles
  dim3 g2(M_ / 256, D_ / 128), b2(256);
  qkv_gemm<<<g2, b2, 0, stream>>>(target, Wh,           WQ_b, QKVh);
  qkv_gemm<<<g2, b2, 0, stream>>>(source, Wh + DD_,     WK_b, QKVh + MN_);
  qkv_gemm<<<g2, b2, 0, stream>>>(memory, Wh + 2 * DD_, WV_b, QKVh + 2 * MN_);

  // K3: per-position head-axis attention (8192 positions, 1 wave each)
  attn_kernel<<<M_ / 8, 256, 0, stream>>>(QKVh, maskp, Ohalf);

  // K4: final per-head projection GEMM, f32 output
  proj_gemm<<<(M_ * H_) / 256, 256, 0, stream>>>(Ohalf, Wph, Wp_b, (float*)d_out);
}
